// Attention_16381005267091
// MI455X (gfx1250) — matrix operand order repaired and checked by reading, not compiled, not run on hardware
//
#include <hip/hip_runtime.h>
#include <hip/hip_bf16.h>

typedef _Float16 v16h __attribute__((ext_vector_type(16)));
typedef _Float16 v8h  __attribute__((ext_vector_type(8)));
typedef float    v8f  __attribute__((ext_vector_type(8)));

#define NTOK 1024
#define CDIM 768
#define NH   12
#define HDIM 64
#define NBATCH 8

#define BM 128
#define BN 64
#define BK 32
#define AST 40   // padded LDS stride (f16 elems) for A tile rows
#define BST 40   // padded LDS stride (f16 elems) for B tile rows (transposed)

enum { OUT_F32 = 0, OUT_Q = 1, OUT_KT = 2, OUT_V = 3, OUT_XHEAD = 4 };

// ---- CDNA5 async global->LDS copy (ASYNCcnt-tracked), 16 bytes per lane ----
__device__ __forceinline__ void async_copy_b128(void* lds_dst, const void* gsrc) {
  unsigned lds_addr = (unsigned)(uintptr_t)lds_dst;               // addr[31:0] = LDS offset
  unsigned long long gaddr = (unsigned long long)(uintptr_t)gsrc;
  asm volatile("global_load_async_to_lds_b128 %0, %1, off"
               :: "v"(lds_addr), "v"(gaddr) : "memory");
}
__device__ __forceinline__ void wait_async0() {
  asm volatile("s_wait_asynccnt 0" ::: "memory");
}

// -------------------- converts --------------------

__global__ void cvt_f32_to_f16(const float* __restrict__ in,
                               _Float16* __restrict__ out, long long n) {
  long long i = (long long)blockIdx.x * 256 + threadIdx.x;
  if (i < n) out[i] = (_Float16)in[i];
}

// WT[k*CDIM + j] = (f16) W[j*CDIM + k]   ->  B layout [K,N], N contiguous
__global__ void transpose_cvt(const float* __restrict__ W,
                              _Float16* __restrict__ WT) {
  int idx = blockIdx.x * 256 + threadIdx.x;     // 768*768 elements
  int k = idx / CDIM;
  int j = idx - k * CDIM;
  WT[idx] = (_Float16)W[j * CDIM + k];
}

// -------------------- generic WMMA GEMM --------------------
// C = A[M,K] (row-major, K contig) * B[K,N] (row-major, N contig)  (+bias[n])
// AF32: A is f32, converted while staging. MODE: epilogue scatter layout.

template<bool AF32, int MODE>
__global__ __launch_bounds__(256)
void gemm_wmma(const void* __restrict__ Av, long long strideA,
               const _Float16* __restrict__ Bg, long long strideB,
               void* __restrict__ Cv, long long strideC, int ldc,
               const float* __restrict__ bias,
               int M, int N, int K, int lda, int ldb) {
  __shared__ __align__(16) _Float16 As[2][BM * AST];
  __shared__ __align__(16) _Float16 Bs[2][BN * BST];

  const int tid  = threadIdx.x;
  const int lane = tid & 31;
  const int w    = tid >> 5;          // 8 waves
  const int wm   = w & 3;             // 4 along M
  const int wn   = w >> 2;            // 2 along N
  const int z    = blockIdx.z;
  const int m0   = blockIdx.y * BM;
  const int n0   = blockIdx.x * BN;

  const _Float16* A16 = (const _Float16*)Av + (AF32 ? 0 : (long long)z * strideA);
  const float*    A32 = (const float*)Av    + (AF32 ? (long long)z * strideA : 0);
  const _Float16* Bb  = Bg + (long long)z * strideB;

  v8f acc[2][2] = {};

  auto stage = [&](int ks, int buf) {
    const int kk = ks * BK;
    // ---- A tile: 128x32 f16, 512 16B chunks, 2 per thread ----
#pragma unroll
    for (int it = 0; it < 2; ++it) {
      int idx = tid + it * 256;
      int row = idx >> 2;
      int kc  = (idx & 3) * 8;
      _Float16* dst = &As[buf][row * AST + kc];
      if constexpr (AF32) {
        const float* src = A32 + (long long)(m0 + row) * lda + kk + kc;
        v8h av;
#pragma unroll
        for (int j = 0; j < 8; ++j) av[j] = (_Float16)src[j];
        *(v8h*)dst = av;
      } else {
        async_copy_b128(dst, A16 + (long long)(m0 + row) * lda + kk + kc);
      }
    }
    // ---- B tile: 32x64 global -> LDS transposed [n][k] ----
    {
      int kr = tid >> 3;
      int nc = (tid & 7) * 8;
      const _Float16* src = Bb + (long long)(kk + kr) * ldb + n0 + nc;
      if (kk + 2 * BK < K)
        __builtin_prefetch(src + 2 * BK * ldb, 0, 1);
      v8h bv = *(const v8h*)src;
#pragma unroll
      for (int j = 0; j < 8; ++j)
        Bs[buf][(nc + j) * BST + kr] = bv[j];
    }
  };

  const int nst = K / BK;
  stage(0, 0);
  for (int ks = 0; ks < nst; ++ks) {
    if constexpr (!AF32) wait_async0();  // our wave's async LDS writes done
    __syncthreads();                     // everyone's writes visible
    if (ks + 1 < nst) stage(ks + 1, (ks + 1) & 1);
    const int buf = ks & 1;

    // A frag: lanes<16 -> kb=0, lanes>=16 -> kb=8; chunks at kb and kb+16
    const int ab = (lane >> 4) << 3;
    // B frag: lanes<16 -> kb=0, lanes>=16 -> kb=8; chunks at kb and kb+16 (stored K-contig)
    const int bb = (lane >> 4) << 3;

    v16h afrag[2], bfrag[2];
#pragma unroll
    for (int mi = 0; mi < 2; ++mi) {
      const _Float16* p = &As[buf][(wm * 32 + mi * 16 + (lane & 15)) * AST + ab];
      v8h lo = *(const v8h*)p;
      v8h hi = *(const v8h*)(p + 16);
#pragma unroll
      for (int j = 0; j < 8; ++j) { afrag[mi][j] = lo[j]; afrag[mi][8 + j] = hi[j]; }
    }
#pragma unroll
    for (int ni = 0; ni < 2; ++ni) {
      const _Float16* p = &Bs[buf][(wn * 32 + ni * 16 + (lane & 15)) * BST + bb];
      v8h lo = *(const v8h*)p;
      v8h hi = *(const v8h*)(p + 16);
#pragma unroll
      for (int j = 0; j < 8; ++j) { bfrag[ni][j] = lo[j]; bfrag[ni][8 + j] = hi[j]; }
    }
#pragma unroll
    for (int mi = 0; mi < 2; ++mi)
#pragma unroll
      for (int ni = 0; ni < 2; ++ni)
        acc[mi][ni] = __builtin_amdgcn_wmma_f32_16x16x32_f16(
            false, afrag[mi], false, bfrag[ni], (short)0, acc[mi][ni], false, false);
  }

  // ---- epilogue: element i of v8f is (M = 8*(lane>=16)+i, N = lane&15) ----
#pragma unroll
  for (int mi = 0; mi < 2; ++mi) {
#pragma unroll
    for (int ni = 0; ni < 2; ++ni) {
      const int mb = m0 + wm * 32 + mi * 16 + ((lane >> 4) << 3);
      const int nn = n0 + wn * 32 + ni * 16 + (lane & 15);
#pragma unroll
      for (int i = 0; i < 8; ++i) {
        const int mm = mb + i;
        float v = acc[mi][ni][i];
        if (bias) v += bias[nn];
        if constexpr (MODE == OUT_F32) {
          float* C = (float*)Cv + (long long)z * strideC;
          C[(long long)mm * ldc + nn] = v;
        } else if constexpr (MODE == OUT_Q || MODE == OUT_V) {
          _Float16* O = (_Float16*)Cv;
          int b = mm >> 10, nt = mm & (NTOK - 1);
          int h = nn >> 6, hd = nn & (HDIM - 1);
          O[((long long)(b * NH + h) * NTOK + nt) * HDIM + hd] = (_Float16)v;
        } else if constexpr (MODE == OUT_KT) {
          _Float16* O = (_Float16*)Cv;
          int b = mm >> 10, nt = mm & (NTOK - 1);
          int h = nn >> 6, hd = nn & (HDIM - 1);
          O[((long long)(b * NH + h) * HDIM + hd) * NTOK + nt] = (_Float16)v;
        } else { // OUT_XHEAD: per (b,h) batch, mm = token, nn = hd
          _Float16* O = (_Float16*)Cv;
          int b = z / NH, h = z - b * NH;
          O[((long long)b * NTOK + mm) * CDIM + h * HDIM + nn] = (_Float16)v;
        }
      }
    }
  }
}

// -------------------- softmax (wave32 per row, scale folded in) ------------

__global__ __launch_bounds__(256)
void softmax_rows(float* __restrict__ attn) {
  const int row  = blockIdx.x * 8 + (threadIdx.x >> 5);
  const int lane = threadIdx.x & 31;
  float* p = attn + (long long)row * NTOK;
  float vals[32];
  float mx = -3.0e30f;
#pragma unroll
  for (int j = 0; j < 32; ++j) {
    vals[j] = p[j * 32 + lane] * 0.125f;  // SCALE = 64^-0.5
    mx = fmaxf(mx, vals[j]);
  }
  for (int off = 16; off; off >>= 1) mx = fmaxf(mx, __shfl_xor(mx, off, 32));
  float s = 0.f;
#pragma unroll
  for (int j = 0; j < 32; ++j) { vals[j] = __expf(vals[j] - mx); s += vals[j]; }
  for (int off = 16; off; off >>= 1) s += __shfl_xor(s, off, 32);
  const float inv = 1.0f / s;
#pragma unroll
  for (int j = 0; j < 32; ++j) p[j * 32 + lane] = vals[j] * inv;
}

// -------------------- launch --------------------

extern "C" void kernel_launch(void* const* d_in, const int* in_sizes, int n_in,
                              void* d_out, int out_size, void* d_ws, size_t ws_size,
                              hipStream_t stream) {
  const float* tfeat = (const float*)d_in[0];
  const float* wq = (const float*)d_in[1];
  const float* bq = (const float*)d_in[2];
  const float* wk = (const float*)d_in[3];
  const float* bk = (const float*)d_in[4];
  const float* wv = (const float*)d_in[5];
  const float* bv = (const float*)d_in[6];
  const float* wp = (const float*)d_in[7];
  const float* bp = (const float*)d_in[8];

  const long long NX = (long long)NBATCH * NTOK * CDIM;   // 6,291,456
  const long long NW = (long long)CDIM * CDIM;            // 589,824

  float* out_x    = (float*)d_out;
  float* out_attn = out_x + NX;                           // [B,H,NT,NT]

  _Float16* ws16 = (_Float16*)d_ws;
  _Float16* x16  = ws16;
  _Float16* wqT  = x16 + NX;
  _Float16* wkT  = wqT + NW;
  _Float16* wvT  = wkT + NW;
  _Float16* wpT  = wvT + NW;
  _Float16* q16  = wpT + NW;     // [B,H,NT,HD]
  _Float16* kT16 = q16 + NX;     // [B,H,HD,NT]
  _Float16* v16  = kT16 + NX;    // [B,H,NT,HD]
  _Float16* xh16 = v16 + NX;     // [B,NT,DIM] head-merged

  dim3 blk(256);

  cvt_f32_to_f16<<<(unsigned)((NX + 255) / 256), blk, 0, stream>>>(tfeat, x16, NX);
  transpose_cvt<<<(unsigned)(NW / 256), blk, 0, stream>>>(wq, wqT);
  transpose_cvt<<<(unsigned)(NW / 256), blk, 0, stream>>>(wk, wkT);
  transpose_cvt<<<(unsigned)(NW / 256), blk, 0, stream>>>(wv, wvT);
  transpose_cvt<<<(unsigned)(NW / 256), blk, 0, stream>>>(wp, wpT);

  const int Mx = NBATCH * NTOK;  // 8192
  // Q/K/V projections: [8192,768] x [768,768]
  dim3 gqkv(CDIM / BN, Mx / BM, 1);
  gemm_wmma<false, OUT_Q><<<gqkv, blk, 0, stream>>>(
      x16, 0, wqT, 0, q16, 0, 0, bq, Mx, CDIM, CDIM, CDIM, CDIM);
  gemm_wmma<false, OUT_KT><<<gqkv, blk, 0, stream>>>(
      x16, 0, wkT, 0, kT16, 0, 0, bk, Mx, CDIM, CDIM, CDIM, CDIM);
  gemm_wmma<false, OUT_V><<<gqkv, blk, 0, stream>>>(
      x16, 0, wvT, 0, v16, 0, 0, bv, Mx, CDIM, CDIM, CDIM, CDIM);

  // scores: per (b,h): [1024,64] x [64,1024] -> attn (pre-softmax)
  dim3 gsc(NTOK / BN, NTOK / BM, NBATCH * NH);
  gemm_wmma<false, OUT_F32><<<gsc, blk, 0, stream>>>(
      q16, (long long)NTOK * HDIM, kT16, (long long)HDIM * NTOK,
      out_attn, (long long)NTOK * NTOK, NTOK, nullptr,
      NTOK, NTOK, HDIM, HDIM, NTOK);

  softmax_rows<<<(NBATCH * NH * NTOK) / 8, blk, 0, stream>>>(out_attn);

  // attn @ V: per (b,h): [1024,1024](f32) x [1024,64] -> head-merged f16
  dim3 gav(HDIM / BN, NTOK / BM, NBATCH * NH);
  gemm_wmma<true, OUT_XHEAD><<<gav, blk, 0, stream>>>(
      out_attn, (long long)NTOK * NTOK, v16, (long long)NTOK * HDIM,
      xh16, 0, 0, nullptr, NTOK, HDIM, NTOK, NTOK, HDIM);

  // output projection: [8192,768] x [768,768] + bp -> out_x
  gemm_wmma<false, OUT_F32><<<gqkv, blk, 0, stream>>>(
      xh16, 0, wpT, 0, out_x, 0, CDIM, bp, Mx, CDIM, CDIM, CDIM, CDIM);
}
